// Attention2D_3075196584135
// MI455X (gfx1250) — matrix operand order repaired and checked by reading, not compiled, not run on hardware
//
#include <hip/hip_runtime.h>
#include <hip/hip_bf16.h>

// ---------------------------------------------------------------------------
// Attention2D for MI455X (gfx1250, wave32, WMMA).
//   1) gn_kernel    : fused double GroupNorm  q -> qn (bf16), vn (bf16)
//   2) proj_kernel  : bf16 WMMA GEMM  Y = W*X + b   (qp scaled by c^-0.5,
//                     kp stored transposed [b,hw,c], vp normal)
//   3) attn_kernel  : flash-attention (online softmax over keys), WMMA for
//                     both K^T*Q scores and V*P, fp32 accumulators
//   4) proj_kernel  : output projection + residual + /sqrt(2)  (fp32 out)
// LDS layouts are chosen so every matrix fragment loads as ds_load_b128
// (16B contiguous per-lane runs), not scalar u16 gathers.
// ---------------------------------------------------------------------------

typedef __attribute__((ext_vector_type(16))) __bf16 v16bf;
typedef __attribute__((ext_vector_type(8)))  float  v8f;

#define B_   8
#define C_   512
#define HW_  4096
#define G_   32

static __device__ inline v8f vzero() {
  v8f z;
#pragma unroll
  for (int i = 0; i < 8; ++i) z[i] = 0.0f;
  return z;
}

static __device__ inline v8f wmma_bf16(v16bf a, v16bf b, v8f c) {
  // D = A(16x32) * B(32x16) + C, fp32 accumulate
  return __builtin_amdgcn_wmma_f32_16x16x32_bf16(
      false, a, false, b, (short)0, c, false, false);
}

// A fragment 16x32 bf16: element(m,k) = base[(row0+m)*ldr + col0 + k]
// Per-lane: one contiguous 16-element k-run (offset hi*16) -> two b128 loads.
static __device__ inline v16bf load_A_g(const __bf16* base, int ldr,
                                        int row0, int col0) {
  int lane = threadIdx.x & 31;
  int m = lane & 15, hi = lane >> 4;
  const __bf16* r = base + (size_t)(row0 + m) * ldr + col0 + hi * 16;
  v16bf a;
#pragma unroll
  for (int e = 0; e < 16; ++e) a[e] = r[e];
  return a;
}

// B fragment 32x16 bf16 from a ROW-MAJOR [k][n] source (strided, scalar).
// Used only for the one-time Q fragment loads from global.
static __device__ inline v16bf load_B_g(const __bf16* base, int ldk,
                                        int row0, int col0) {
  int lane = threadIdx.x & 31;
  int n = lane & 15, hi = lane >> 4;
  v16bf b;
#pragma unroll
  for (int e = 0; e < 16; ++e)
    b[e] = base[(size_t)(row0 + e + hi * 16) * ldk + col0 + n];
  return b;
}

// B fragment 32x16 bf16 from a TRANSPOSED [n][k] source: lane n reads a
// contiguous 16-element k-run (offset hi*16) -> two b128 loads.
static __device__ inline v16bf load_B_T(const __bf16* baseT, int ldn,
                                        int n0) {
  int lane = threadIdx.x & 31;
  int n = lane & 15, hi = lane >> 4;
  const __bf16* r = baseT + (size_t)(n0 + n) * ldn + hi * 16;
  v16bf b;
#pragma unroll
  for (int e = 0; e < 16; ++e) b[e] = r[e];
  return b;
}

static __device__ inline float block_sum(float v, float* buf) {
  int t = threadIdx.x;
  __syncthreads();
  buf[t] = v;
  __syncthreads();
  for (int s = 128; s; s >>= 1) {
    if (t < s) buf[t] += buf[t + s];
    __syncthreads();
  }
  return buf[0];
}

// ---------------------------------------------------------------------------
// Kernel 1: fused double GroupNorm. One block per (batch, group); a group is
// 16 channels * 4096 = 64K contiguous floats (256KB, L2-resident between the
// three streaming passes).
// ---------------------------------------------------------------------------
__global__ __launch_bounds__(256) void gn_kernel(
    const float* __restrict__ q, const float* __restrict__ gamma,
    const float* __restrict__ beta, __bf16* __restrict__ qn,
    __bf16* __restrict__ vn) {
  __shared__ float sred[256];
  int b = blockIdx.x >> 5, g = blockIdx.x & 31, tid = threadIdx.x;
  size_t base = ((size_t)b * C_ + g * 16) * HW_;
  const float inv = 1.0f / 65536.0f;

  float s = 0.f, ss = 0.f;
  for (int j = 0; j < 256; ++j) {
    float v = q[base + tid + j * 256];
    s += v; ss += v * v;
  }
  s = block_sum(s, sred);
  ss = block_sum(ss, sred);
  float mu = s * inv;
  float rs = rsqrtf(ss * inv - mu * mu + 1e-6f);

  float s2 = 0.f, ss2 = 0.f;
  for (int j = 0; j < 256; ++j) {
    int l = tid + j * 256;
    int ch = g * 16 + (l >> 12);
    float qv = (q[base + l] - mu) * rs * gamma[ch] + beta[ch];
    qn[base + l] = (__bf16)qv;
    s2 += qv; ss2 += qv * qv;
  }
  s2 = block_sum(s2, sred);
  ss2 = block_sum(ss2, sred);
  float mu2 = s2 * inv;
  float rs2 = rsqrtf(ss2 * inv - mu2 * mu2 + 1e-6f);

  for (int j = 0; j < 256; ++j) {
    int l = tid + j * 256;
    int ch = g * 16 + (l >> 12);
    float qv = (q[base + l] - mu) * rs * gamma[ch] + beta[ch];
    float vv = (qv - mu2) * rs2 * gamma[ch] + beta[ch];
    vn[base + l] = (__bf16)vv;
  }
}

// ---------------------------------------------------------------------------
// Kernel 2/4: 1x1-conv GEMM  Y[o,n] = sum_c W[o,c] X[c,n] + bias[o].
// Block tile 64(M) x 128(N), 8 waves as 2x4, each wave 2x2 WMMA tiles.
// W tile staged row-major [64][32] (A frags -> b128 loads); X tile staged
// TRANSPOSED [128][32] so B frags are contiguous k-runs (b128 loads).
// Modes: bf16 store (optionally scaled / transposed) or fp32 residual store.
// ---------------------------------------------------------------------------
__global__ __launch_bounds__(256) void proj_kernel(
    const float* __restrict__ W, const __bf16* __restrict__ X,
    const float* __restrict__ bias, float scale, __bf16* __restrict__ outB,
    int transposeOut, const float* __restrict__ resid,
    float* __restrict__ outF) {
  __shared__ __attribute__((aligned(16))) __bf16 Wlds[64 * 32];
  __shared__ __attribute__((aligned(16))) __bf16 XldsT[128 * 32];
  int tid = threadIdx.x, wid = tid >> 5, lane = tid & 31;
  int m0 = blockIdx.y * 64;
  int n0 = blockIdx.x * 128;
  int bb = n0 >> 12;      // batch of this n-tile (4096 % 128 == 0)
  int i0 = n0 & 4095;     // pixel offset within batch
  const __bf16* Xb = X + (size_t)bb * C_ * HW_ + i0;  // X[c][i]
  int wm = (wid >> 2) * 32;
  int wn = (wid & 3) * 32;

  v8f acc[2][2];
#pragma unroll
  for (int mt = 0; mt < 2; ++mt)
#pragma unroll
    for (int nt = 0; nt < 2; ++nt) acc[mt][nt] = vzero();

  // per-thread 4x4 transpose block for X staging
  int tn = (tid & 31) * 4;   // n offset, 0..124
  int tk = (tid >> 5) * 4;   // k offset, 0..28

  for (int k0 = 0; k0 < C_; k0 += 32) {
    {  // stage W tile 64x32 fp32 -> bf16 (W is 1MB, L2-resident)
      int l = tid * 8, row = l >> 5, col = l & 31;
#pragma unroll
      for (int j = 0; j < 8; ++j)
        Wlds[row * 32 + col + j] =
            (__bf16)W[(size_t)(m0 + row) * C_ + k0 + col + j];
    }
    {  // stage X tile 32(k) x 128(n), transposed into XldsT[n][k]
      __bf16 t4[4][4];
#pragma unroll
      for (int kk = 0; kk < 4; ++kk) {
        const __bf16* src = Xb + (size_t)(k0 + tk + kk) * HW_ + tn;
#pragma unroll
        for (int nn = 0; nn < 4; ++nn) t4[kk][nn] = src[nn];
      }
#pragma unroll
      for (int nn = 0; nn < 4; ++nn)
#pragma unroll
        for (int kk = 0; kk < 4; ++kk)
          XldsT[(tn + nn) * 32 + tk + kk] = t4[kk][nn];
    }
    __syncthreads();
    v16bf af[2], bf[2];
#pragma unroll
    for (int mt = 0; mt < 2; ++mt) af[mt] = load_A_g(Wlds, 32, wm + mt * 16, 0);
#pragma unroll
    for (int nt = 0; nt < 2; ++nt) bf[nt] = load_B_T(XldsT, 32, wn + nt * 16);
#pragma unroll
    for (int mt = 0; mt < 2; ++mt)
#pragma unroll
      for (int nt = 0; nt < 2; ++nt)
        acc[mt][nt] = wmma_bf16(af[mt], bf[nt], acc[mt][nt]);
    __syncthreads();
  }

  int hi = lane >> 4, nl = lane & 15;
#pragma unroll
  for (int mt = 0; mt < 2; ++mt)
#pragma unroll
    for (int nt = 0; nt < 2; ++nt)
#pragma unroll
      for (int r = 0; r < 8; ++r) {
        int o = m0 + wm + mt * 16 + r + hi * 8;
        int ni = i0 + wn + nt * 16 + nl;
        float v = (acc[mt][nt][r] + bias[o]) * scale;
        if (outF) {
          size_t idx = ((size_t)bb * C_ + o) * HW_ + ni;
          outF[idx] = (v + resid[idx]) * 0.70710678118654752f;  // /sqrt(2)
        } else if (transposeOut) {
          outB[((size_t)bb * HW_ + ni) * C_ + o] = (__bf16)v;
        } else {
          outB[((size_t)bb * C_ + o) * HW_ + ni] = (__bf16)v;
        }
      }
}

// ---------------------------------------------------------------------------
// Kernel 3: flash attention. One block per (batch, 32-query tile).
// 8 waves; each wave owns a 64-channel slab of the V*P accumulator (4x2 WMMA
// tiles, fp32). Score partials (per-wave over 64 channels) reduced in LDS;
// online softmax stats per query column write the probability tile directly
// as a transposed bf16 tile PT[y][x] so P fragments load as b128; keys are
// streamed in 32-wide tiles (Q/K/V all L2-resident: 96MB < 192MB L2).
// ---------------------------------------------------------------------------
__global__ __launch_bounds__(256) void attn_kernel(
    const __bf16* __restrict__ qp, const __bf16* __restrict__ kpT,
    const __bf16* __restrict__ vp, __bf16* __restrict__ outa) {
  __shared__ float Spart[8][32 * 32];                       // 32KB
  __shared__ float Sred[32 * 32];                           // 4KB
  __shared__ __attribute__((aligned(16))) __bf16 PT[32 * 32];  // 2KB, [y][x]
  __shared__ float mbuf[32], lbuf[32], albuf[32];

  int tid = threadIdx.x, wid = tid >> 5, lane = tid & 31;
  int hi = lane >> 4, nl = lane & 15;
  int b = blockIdx.x >> 7;
  int y0 = (blockIdx.x & 127) * 32;
  int cw = wid * 64;  // channel slab of this wave

  const __bf16* qpb = qp + (size_t)b * C_ * HW_;
  const __bf16* ktb = kpT + (size_t)b * HW_ * C_;
  const __bf16* vpb = vp + (size_t)b * C_ * HW_;

  // Q fragments (held in registers for all 128 key tiles)
  v16bf Qf[2][2];
#pragma unroll
  for (int cs = 0; cs < 2; ++cs)
#pragma unroll
    for (int ys = 0; ys < 2; ++ys)
      Qf[cs][ys] = load_B_g(qpb, HW_, cw + cs * 32, y0 + ys * 16);

  v8f acc[4][2];
#pragma unroll
  for (int mt = 0; mt < 4; ++mt)
#pragma unroll
    for (int ys = 0; ys < 2; ++ys) acc[mt][ys] = vzero();

  for (int xt = 0; xt < 128; ++xt) {
    int x0 = xt * 32;

    // --- partial scores S[x,y] over this wave's 64 channels: A = K^T ---
    v16bf Ak[2][2];
#pragma unroll
    for (int xs = 0; xs < 2; ++xs)
#pragma unroll
      for (int cs = 0; cs < 2; ++cs)
        Ak[xs][cs] = load_A_g(ktb, C_, x0 + xs * 16, cw + cs * 32);
#pragma unroll
    for (int xs = 0; xs < 2; ++xs)
#pragma unroll
      for (int ys = 0; ys < 2; ++ys) {
        v8f s = vzero();
#pragma unroll
        for (int cs = 0; cs < 2; ++cs) s = wmma_bf16(Ak[xs][cs], Qf[cs][ys], s);
#pragma unroll
        for (int r = 0; r < 8; ++r)
          Spart[wid][(xs * 16 + r + hi * 8) * 32 + (ys * 16 + nl)] = s[r];
      }
    __syncthreads();

    // --- reduce the 8 per-wave partials ---
#pragma unroll
    for (int u = 0; u < 4; ++u) {
      int e = tid * 4 + u;
      float v = 0.f;
#pragma unroll
      for (int w = 0; w < 8; ++w) v += Spart[w][e];
      Sred[e] = v;
    }
    __syncthreads();

    // --- online softmax stats; emit bf16 probs transposed: PT[y][x] ---
    if (tid < 32) {
      int y = tid;
      float mold = (xt == 0) ? -3.0e38f : mbuf[y];
      float lold = (xt == 0) ? 0.f : lbuf[y];
      float tm = mold;
      for (int x = 0; x < 32; ++x) tm = fmaxf(tm, Sred[x * 32 + y]);
      float al = __expf(mold - tm);
      float ps = 0.f;
      for (int x = 0; x < 32; ++x) {
        float p = __expf(Sred[x * 32 + y] - tm);
        PT[y * 32 + x] = (__bf16)p;  // contiguous 64B store per thread
        ps += p;
      }
      mbuf[y] = tm;
      lbuf[y] = lold * al + ps;
      albuf[y] = al;
    }
    __syncthreads();

    // --- rescale accumulator, then acc += V * P ---
    float al0 = albuf[nl], al1 = albuf[16 + nl];
#pragma unroll
    for (int mt = 0; mt < 4; ++mt)
#pragma unroll
      for (int r = 0; r < 8; ++r) {
        acc[mt][0][r] *= al0;
        acc[mt][1][r] *= al1;
      }
    v16bf Pf[2];
#pragma unroll
    for (int ys = 0; ys < 2; ++ys) Pf[ys] = load_B_T(PT, 32, ys * 16);
#pragma unroll
    for (int mt = 0; mt < 4; ++mt) {
      v16bf Vf = load_A_g(vpb, HW_, cw + mt * 16, x0);
#pragma unroll
      for (int ys = 0; ys < 2; ++ys)
        acc[mt][ys] = wmma_bf16(Vf, Pf[ys], acc[mt][ys]);
    }

    // prefetch next K/V tiles (global_prefetch_b8; L2 hits feed the loop)
    if (xt < 127) {
      __builtin_prefetch(ktb + (size_t)(x0 + 32 + lane) * C_ + cw, 0, 1);
      __builtin_prefetch(vpb + (size_t)(cw + lane) * HW_ + x0 + 32, 0, 1);
    }
  }

  // --- normalize and store ---
  __syncthreads();
  if (tid < 32) albuf[tid] = 1.0f / lbuf[tid];
  __syncthreads();
  float li0 = albuf[nl], li1 = albuf[16 + nl];
#pragma unroll
  for (int mt = 0; mt < 4; ++mt)
#pragma unroll
    for (int ys = 0; ys < 2; ++ys)
#pragma unroll
      for (int r = 0; r < 8; ++r) {
        int c = cw + mt * 16 + r + hi * 8;
        int y = y0 + ys * 16 + nl;
        float v = acc[mt][ys][r] * (ys ? li1 : li0);
        outa[((size_t)b * C_ + c) * HW_ + y] = (__bf16)v;
      }
}

// ---------------------------------------------------------------------------
extern "C" void kernel_launch(void* const* d_in, const int* in_sizes, int n_in,
                              void* d_out, int out_size, void* d_ws,
                              size_t ws_size, hipStream_t stream) {
  (void)in_sizes; (void)n_in; (void)out_size; (void)ws_size;
  const float* q     = (const float*)d_in[0];
  const float* gamma = (const float*)d_in[1];
  const float* beta  = (const float*)d_in[2];
  const float* wq    = (const float*)d_in[3];
  const float* bq    = (const float*)d_in[4];
  const float* wk    = (const float*)d_in[5];
  const float* bk    = (const float*)d_in[6];
  const float* wv    = (const float*)d_in[7];
  const float* bv    = (const float*)d_in[8];
  const float* wo    = (const float*)d_in[9];
  const float* bo    = (const float*)d_in[10];
  float* out = (float*)d_out;

  char* ws = (char*)d_ws;
  const size_t SZ = (size_t)B_ * C_ * HW_ * 2;  // 32MB per bf16 tensor
  __bf16* qn = (__bf16*)(ws + 0 * SZ);
  __bf16* vn = (__bf16*)(ws + 1 * SZ);
  __bf16* qp = (__bf16*)(ws + 2 * SZ);
  __bf16* kt = (__bf16*)(ws + 3 * SZ);
  __bf16* vp = (__bf16*)(ws + 4 * SZ);
  __bf16* ao = qn;  // qn dead after qp projection -> reuse for attn output

  dim3 blk(256);
  dim3 pg(B_ * HW_ / 128, C_ / 64);
  const float scale = 0.044194173824159216f;  // 512^-0.5 folded into qp

  gn_kernel<<<dim3(B_ * G_), blk, 0, stream>>>(q, gamma, beta, qn, vn);
  proj_kernel<<<pg, blk, 0, stream>>>(wq, qn, bq, scale, qp, 0, nullptr, nullptr);
  proj_kernel<<<pg, blk, 0, stream>>>(wk, vn, bk, 1.0f, kt, 1, nullptr, nullptr);
  proj_kernel<<<pg, blk, 0, stream>>>(wv, vn, bv, 1.0f, vp, 0, nullptr, nullptr);
  attn_kernel<<<dim3(B_ * (HW_ / 32)), blk, 0, stream>>>(qp, kt, vp, ao);
  proj_kernel<<<pg, blk, 0, stream>>>(wo, ao, bo, 1.0f, nullptr, 0, q, out);
}
